// GATv2Layer_10823317586432
// MI455X (gfx1250) — matrix operand order repaired and checked by reading, not compiled, not run on hardware
//
#include <hip/hip_runtime.h>

typedef _Float16 v16h __attribute__((ext_vector_type(16)));
typedef float    v8f  __attribute__((ext_vector_type(8)));
typedef float    v4f  __attribute__((ext_vector_type(4)));

#define NN        8192
#define FDIM      128
#define NEG_SLOPE 0.2f

// ---------------------------------------------------------------------------
// Phase 1: Wh = h @ W^T ; s1 = Wh·a1 ; s2 = Wh·a2 ; and store Wh as f16 in
// the exact CDNA5 WMMA B-operand layout (16x16x32 f16: B is 32x16 per tile,
// lane = (n&15) + 16*((k>>3)&1), halfword pos = (k&7) + 8*(k>>4)) so phase 2 can load each
// lane's v16h as one contiguous 32-byte read.
// ---------------------------------------------------------------------------
__global__ __launch_bounds__(128)
void gat_wh_kernel(const float* __restrict__ h, const float* __restrict__ W,
                   const float* __restrict__ a,
                   float* __restrict__ s1, float* __restrict__ s2,
                   _Float16* __restrict__ blay) {
  __shared__ float hrow[FDIM];
  __shared__ float r1[FDIM];
  __shared__ float r2[FDIM];
  const int i = blockIdx.x;     // row of h
  const int t = threadIdx.x;    // output feature f

  hrow[t] = h[(size_t)i * FDIM + t];
  __syncthreads();

  const v4f* wr  = (const v4f*)(W + (size_t)t * FDIM);
  const v4f* hr4 = (const v4f*)hrow;
  float wh = 0.f;
#pragma unroll
  for (int k4 = 0; k4 < FDIM / 4; ++k4) {
    v4f w4 = wr[k4];
    v4f h4 = hr4[k4];
    wh = fmaf(h4.x, w4.x, wh);
    wh = fmaf(h4.y, w4.y, wh);
    wh = fmaf(h4.z, w4.z, wh);
    wh = fmaf(h4.w, w4.w, wh);
  }

  // Scatter Wh[i][t] into WMMA-B layout: j-tile = i>>5, k = i&31, n-tile = t>>4
  {
    const int k    = i & 31;
    const int jt   = i >> 5;
    const int nt   = t >> 4;
    const int n    = t & 15;
    const int lane = n + 16 * ((k >> 3) & 1);
    const int pos  = (k & 7) + 8 * (k >> 4);
    blay[(((size_t)(jt * 8 + nt)) * 32 + lane) * 16 + pos] = (_Float16)wh;
  }

  // Row scalars s1[i], s2[i]
  r1[t] = wh * a[t];
  r2[t] = wh * a[FDIM + t];
  __syncthreads();
#pragma unroll
  for (int s = 64; s > 0; s >>= 1) {
    if (t < s) { r1[t] += r1[t + s]; r2[t] += r2[t + s]; }
    __syncthreads();
  }
  if (t == 0) { s1[i] = r1[0]; s2[i] = r2[0]; }
}

// ---------------------------------------------------------------------------
// Phase 2: flash-style fused attention. One wave owns 16 output rows.
// Streams adj once (HBM-bound 256 MB, non-temporal), builds the un-normalized
// P tile (exp of leaky-relu score * adj) directly in the ISA A-operand lane
// layout (no cross-lane exchange needed), accumulates row sums Z and P @ Wh
// with v_wmma_f32_16x16x32_f16, then normalizes on store.
//
// A-layout (16x32 f16): lane = m + 16*((k>>3)&1), pos = (k&7) + 8*(k>>4).
// => lane (m,hi) owns k in {8hi..8hi+7} u {16+8hi..23+8hi}; filling pos 0..7
// from the first group and pos 8..15 from the second needs no shuffles.
// ---------------------------------------------------------------------------
__global__ __launch_bounds__(64)
void gat_attn_kernel(const float* __restrict__ adj,
                     const float* __restrict__ s1v,
                     const float* __restrict__ s2v,
                     const v16h* __restrict__ blay,
                     float* __restrict__ out) {
  const int lane = threadIdx.x & 31;
  const int wave = threadIdx.x >> 5;
  const int wid  = blockIdx.x * 2 + wave;   // 512 waves total
  const int i0   = wid * 16;
  const int m    = lane & 15;               // row within the 16-row stripe
  const int hi   = lane >> 4;               // which k-subset this lane owns

  const float s1m = s1v[i0 + m];

  v8f acc[8];
  {
    v8f vz = {0.f, 0.f, 0.f, 0.f, 0.f, 0.f, 0.f, 0.f};
#pragma unroll
    for (int nt = 0; nt < 8; ++nt) acc[nt] = vz;
  }
  float z = 0.f;

  // Lane base: column j0 + 8*hi of row i0+m. The lane pair (m,0)/(m,1)
  // covers 128 contiguous bytes of the row => fully coalesced b128 loads.
  const float* adjrow = adj + (size_t)(i0 + m) * NN + hi * 8;
  const float* s2row  = s2v + hi * 8;

  union F16U { v4f f4[4]; float f[16]; };
  union AUU  { v16h v; _Float16 hh[16]; };

  for (int tile = 0; tile < NN / 32; ++tile) {
    const int j0 = tile * 32;

    // k groups: pos 0..7 -> cols j0+8hi+0..7 ; pos 8..15 -> cols j0+16+8hi+0..7
    const v4f* ap = (const v4f*)(adjrow + j0);
    const v4f* sp = (const v4f*)(s2row + j0);
    F16U au, su;
    au.f4[0] = __builtin_nontemporal_load(&ap[0]);   // single-use 256MB stream:
    au.f4[1] = __builtin_nontemporal_load(&ap[1]);   // keep it out of L2 so the
    au.f4[2] = __builtin_nontemporal_load(&ap[4]);   // reused blay/s2 stay hot
    au.f4[3] = __builtin_nontemporal_load(&ap[5]);
    su.f4[0] = sp[0];
    su.f4[1] = sp[1];
    su.f4[2] = sp[4];
    su.f4[3] = sp[5];
    if (tile + 1 < NN / 32) __builtin_prefetch(adjrow + j0 + 32, 0, 0);

    // 16 scores: e = leaky(s_i + s_j); masked = e*adj; p = exp(masked).
    // adj==0 => p = exp(0) = 1, exactly matching the reference softmax.
    AUU A;
#pragma unroll
    for (int kk = 0; kk < 16; ++kk) {
      float ts = s1m + su.f[kk];
      float lr = fmaxf(ts, 0.f) + NEG_SLOPE * fminf(ts, 0.f);
      float e  = lr * au.f[kk];
      float p  = __expf(e);
      z += p;
      A.hh[kk] = (_Float16)p;   // pos kk directly: no cross-lane exchange
    }

    // 8 WMMAs covering the full F=128 output width.
    const v16h* bt = blay + (size_t)tile * 8 * 32 + lane;
#pragma unroll
    for (int nt = 0; nt < 8; ++nt) {
      v16h b = bt[(size_t)nt * 32];   // contiguous 32B per lane (pre-swizzled)
      acc[nt] = __builtin_amdgcn_wmma_f32_16x16x32_f16(
          false, A.v, false, b, (short)0, acc[nt], false, false);
    }
  }

  // Z for row i0+m is split across the lane pair (m, m+16).
  z += __shfl_xor(z, 16);

  // C/D layout: VGPR r -> row r (lanes 0-15) / r+8 (lanes 16-31), col = m.
#pragma unroll
  for (int r = 0; r < 8; ++r) {
    const int   rr  = hi ? (r + 8) : r;
    const float zr  = __shfl(z, rr);       // lanes 0..15 hold Z per row
    const float inv = 1.0f / zr;
    const size_t row = (size_t)(i0 + rr);
#pragma unroll
    for (int nt = 0; nt < 8; ++nt)
      out[row * FDIM + nt * 16 + m] = acc[nt][r] * inv;
  }
}

// ---------------------------------------------------------------------------
extern "C" void kernel_launch(void* const* d_in, const int* in_sizes, int n_in,
                              void* d_out, int out_size, void* d_ws, size_t ws_size,
                              hipStream_t stream) {
  const float* h   = (const float*)d_in[0];   // [8192,128]
  const float* adj = (const float*)d_in[1];   // [8192,8192]
  const float* W   = (const float*)d_in[2];   // [128,128]
  const float* a   = (const float*)d_in[3];   // [1,256]
  float*       out = (float*)d_out;           // [8192,128]

  char*     ws   = (char*)d_ws;
  float*    s1   = (float*)ws;                        // 8192 f32
  float*    s2   = (float*)(ws + NN * sizeof(float)); // 8192 f32
  _Float16* blay = (_Float16*)(ws + 2 * NN * sizeof(float)); // 8192*128 f16 (2 MB)

  gat_wh_kernel<<<NN, FDIM, 0, stream>>>(h, W, a, s1, s2, blay);
  gat_attn_kernel<<<NN / 32, 64, 0, stream>>>(adj, s1, s2, (const v16h*)blay, out);
}
